// PhysicsInformedGNN_42236708389002
// MI455X (gfx1250) — compile-verified, matrix operand order checked by reading, not run on hardware
//
#include <hip/hip_runtime.h>
#include <hip/hip_bf16.h>
#include <math.h>

#define FIN   10
#define HEADS 4
#define CCH   64
#define HCD   256
#define GRP   64
#define NCLS  8

typedef __attribute__((ext_vector_type(16))) _Float16 v16h;
typedef __attribute__((ext_vector_type(8)))  float    v8f;

// ---- monotonic float<->uint encoding for atomicMax on floats ----
__device__ __forceinline__ unsigned fenc(float f) {
  unsigned u = __float_as_uint(f);
  return (u & 0x80000000u) ? ~u : (u | 0x80000000u);
}
__device__ __forceinline__ float fdec(unsigned u) {
  return (u & 0x80000000u) ? __uint_as_float(u & 0x7fffffffu) : __uint_as_float(~u);
}
#define FENC_NEG_INF 0x007FFFFFu  // fenc(-inf)

// edge idx < E -> edge_index pair; idx >= E -> self loop (idx-E, idx-E)
__device__ __forceinline__ void edge_sd(const int* __restrict__ ei, int idx, int E,
                                        int& s, int& d) {
  if (idx < E) { s = ei[idx]; d = ei[E + idx]; }
  else         { s = idx - E; d = s; }
}

// ----------------------------- fills -----------------------------
__global__ void k_fill_f32(float* p, float v, size_t n) {
  size_t i = (size_t)blockIdx.x * blockDim.x + threadIdx.x;
  if (i < n) p[i] = v;
}
__global__ void k_fill_u32(unsigned* p, unsigned v, size_t n) {
  size_t i = (size_t)blockIdx.x * blockDim.x + threadIdx.x;
  if (i < n) p[i] = v;
}

// ------------------------- layer-1 linear (K=10, scalar) -------------------------
__global__ void k_lin1(const float* __restrict__ x, const float* __restrict__ W,
                       float* __restrict__ h, int N) {
  size_t t = (size_t)blockIdx.x * blockDim.x + threadIdx.x;
  if (t >= (size_t)N * HCD) return;
  int n = (int)(t >> 8), j = (int)(t & 255);
  const float* xr = x + (size_t)n * FIN;
  float acc = 0.f;
#pragma unroll
  for (int k = 0; k < FIN; ++k) acc += xr[k] * W[k * HCD + j];
  h[t] = acc;
}

// ------------------------- attention source/dest terms -------------------------
__global__ void k_esed(const float* __restrict__ h, const float* __restrict__ as_,
                       const float* __restrict__ ad_, float* __restrict__ es,
                       float* __restrict__ ed, int N) {
  size_t t = (size_t)blockIdx.x * blockDim.x + threadIdx.x;
  if (t >= (size_t)N * HEADS) return;
  int n = (int)(t >> 2), hd = (int)(t & 3);
  const float* hr = h + (size_t)n * HCD + hd * CCH;
  float a = 0.f, b = 0.f;
#pragma unroll 8
  for (int c = 0; c < CCH; ++c) { a += hr[c] * as_[hd * CCH + c]; b += hr[c] * ad_[hd * CCH + c]; }
  es[t] = a; ed[t] = b;
}

// ------------------------- edge softmax (3 passes) -------------------------
__global__ void k_edge_max(const int* __restrict__ ei, int E, int E2,
                           const float* __restrict__ es, const float* __restrict__ ed,
                           float* __restrict__ ee, unsigned* __restrict__ mEnc) {
  size_t t = (size_t)blockIdx.x * blockDim.x + threadIdx.x;
  if (t >= (size_t)E2 * HEADS) return;
  int edge = (int)(t >> 2), hd = (int)(t & 3), s, d;
  edge_sd(ei, edge, E, s, d);
  float v = es[(size_t)s * HEADS + hd] + ed[(size_t)d * HEADS + hd];
  v = v > 0.f ? v : 0.2f * v;  // leaky_relu(0.2)
  ee[t] = v;
  atomicMax(mEnc + (size_t)d * HEADS + hd, fenc(v));
}

__global__ void k_edge_exp(const int* __restrict__ ei, int E, int E2,
                           float* __restrict__ ee, const unsigned* __restrict__ mEnc,
                           float* __restrict__ ssum) {
  size_t t = (size_t)blockIdx.x * blockDim.x + threadIdx.x;
  if (t >= (size_t)E2 * HEADS) return;
  int edge = (int)(t >> 2), hd = (int)(t & 3), s, d;
  edge_sd(ei, edge, E, s, d);
  float ex = expf(ee[t] - fdec(mEnc[(size_t)d * HEADS + hd]));
  ee[t] = ex;
  atomicAdd(ssum + (size_t)d * HEADS + hd, ex);
}

__global__ void k_edge_aggr(const int* __restrict__ ei, int E, int E2,
                            const float* __restrict__ ee, const float* __restrict__ ssum,
                            const float* __restrict__ h, float* __restrict__ aggr) {
  size_t t = (size_t)blockIdx.x * blockDim.x + threadIdx.x;
  if (t >= (size_t)E2 * HCD) return;
  int edge = (int)(t >> 8), c = (int)(t & 255), hd = c >> 6, s, d;
  edge_sd(ei, edge, E, s, d);
  float alpha = ee[(size_t)edge * HEADS + hd] / (ssum[(size_t)d * HEADS + hd] + 1e-16f);
  atomicAdd(aggr + (size_t)d * HCD + c, h[(size_t)s * HCD + c] * alpha);
}

// ------------------------- bias + BN + ELU (+f16 copy for WMMA) -------------------------
__global__ void k_post(const float* __restrict__ aggr, const float* __restrict__ bi,
                       const float* __restrict__ g, const float* __restrict__ b,
                       const float* __restrict__ m, const float* __restrict__ v,
                       float* __restrict__ outF, _Float16* __restrict__ outH, int N) {
  size_t t = (size_t)blockIdx.x * blockDim.x + threadIdx.x;
  if (t >= (size_t)N * HCD) return;
  int c = (int)(t & 255);
  float val = aggr[t] + bi[c];
  val = (val - m[c]) * rsqrtf(v[c] + 1e-5f) * g[c] + b[c];
  val = val > 0.f ? val : (expf(val) - 1.f);
  outF[t] = val;
  outH[t] = (_Float16)val;
}

// ------------------------- weight transpose+cast: Bt[n][k] = (f16)W[k][n] -------------------------
__global__ void k_wT(const float* __restrict__ W, _Float16* __restrict__ Bt, int K, int Ncol) {
  size_t t = (size_t)blockIdx.x * blockDim.x + threadIdx.x;
  if (t >= (size_t)K * Ncol) return;
  int k = (int)(t / Ncol), n = (int)(t % Ncol);
  Bt[(size_t)n * K + k] = (_Float16)W[t];
}

// ------------------------- WMMA GEMM: D[M,Ncol] = A[M,K](f16) x Bt[Ncol,K]^T(f16) -------------
// One wave per 16x64 output strip: 4 accumulators share one A fragment per K-step,
// issuing 4 back-to-back v_wmma per A load (4x better A reuse than 16x16/wave).
// Requires: M % 16 == 0, K % 32 == 0, Ncol % 64 == 0.
__global__ void k_gemm_wmma(const _Float16* __restrict__ A, const _Float16* __restrict__ Bt,
                            float* __restrict__ D, int M, int K, int Ncol) {
  int wave = (int)((blockIdx.x * blockDim.x + threadIdx.x) >> 5);
  int lane = (int)(threadIdx.x & 31);
  int mt = M >> 4, ntB = Ncol >> 6;                 // 64-wide N strips
  if (wave >= mt * ntB) return;                     // wave-uniform: EXEC all-ones at WMMA
  int tm = wave / ntB, tnB = wave % ntB;
  int laneLo = lane & 15, laneHi = lane >> 4;
  const _Float16* Arow = A + (size_t)(tm * 16 + laneLo) * K;
  const _Float16* B0   = Bt + (size_t)(tnB * 64 + laneLo) * K;
  const int kgrp = laneHi * 8;
  v8f c0 = {}, c1 = {}, c2 = {}, c3 = {};
  for (int k0 = 0; k0 < K; k0 += 32) {
    v16h a, b0, b1, b2, b3;
#pragma unroll
    for (int v = 0; v < 8; ++v) {
      int kk = k0 + ((v & 3) * 2) + ((v >> 2) * 16) + kgrp;  // ISA 16-bit A/B 16x32 layout
      a[2 * v]      = Arow[kk];
      a[2 * v + 1]  = Arow[kk + 1];
      b0[2 * v]     = B0[kk];
      b0[2 * v + 1] = B0[kk + 1];
      b1[2 * v]     = B0[(size_t)16 * K + kk];
      b1[2 * v + 1] = B0[(size_t)16 * K + kk + 1];
      b2[2 * v]     = B0[(size_t)32 * K + kk];
      b2[2 * v + 1] = B0[(size_t)32 * K + kk + 1];
      b3[2 * v]     = B0[(size_t)48 * K + kk];
      b3[2 * v + 1] = B0[(size_t)48 * K + kk + 1];
    }
    c0 = __builtin_amdgcn_wmma_f32_16x16x32_f16(false, a, false, b0, (short)0, c0, false, false);
    c1 = __builtin_amdgcn_wmma_f32_16x16x32_f16(false, a, false, b1, (short)0, c1, false, false);
    c2 = __builtin_amdgcn_wmma_f32_16x16x32_f16(false, a, false, b2, (short)0, c2, false, false);
    c3 = __builtin_amdgcn_wmma_f32_16x16x32_f16(false, a, false, b3, (short)0, c3, false, false);
  }
  int mBase = tm * 16 + laneHi * 8;
#pragma unroll
  for (int r = 0; r < 8; ++r) {
    float* drow = D + (size_t)(mBase + r) * Ncol + tnB * 64 + laneLo;
    drow[0]  = c0[r];
    drow[16] = c1[r];
    drow[32] = c2[r];
    drow[48] = c3[r];
  }
}

// ------------------------- GCN pieces -------------------------
__global__ void k_deg(const int* __restrict__ ei, int E, int E2, float* __restrict__ deg) {
  size_t t = (size_t)blockIdx.x * blockDim.x + threadIdx.x;
  if (t >= (size_t)E2) return;
  int s, d; edge_sd(ei, (int)t, E, s, d);
  atomicAdd(deg + d, 1.0f);
}
__global__ void k_dinv(const float* __restrict__ deg, float* __restrict__ dinv, int N) {
  size_t t = (size_t)blockIdx.x * blockDim.x + threadIdx.x;
  if (t < (size_t)N) dinv[t] = rsqrtf(fmaxf(deg[t], 1.0f));
}
__global__ void k_gcn_edge(const int* __restrict__ ei, int E, int E2,
                           const float* __restrict__ xw, const float* __restrict__ dinv,
                           float* __restrict__ out) {
  size_t t = (size_t)blockIdx.x * blockDim.x + threadIdx.x;
  if (t >= (size_t)E2 * CCH) return;
  int edge = (int)(t >> 6), c = (int)(t & 63), s, d;
  edge_sd(ei, edge, E, s, d);
  atomicAdd(out + (size_t)d * CCH + c, xw[(size_t)s * CCH + c] * dinv[s] * dinv[d]);
}
__global__ void k_post3(const float* __restrict__ in, const float* __restrict__ bg,
                        const float* __restrict__ g, const float* __restrict__ b,
                        const float* __restrict__ m, const float* __restrict__ v,
                        float* __restrict__ hOut, int N) {
  size_t t = (size_t)blockIdx.x * blockDim.x + threadIdx.x;
  if (t >= (size_t)N * CCH) return;
  int c = (int)(t & 63);
  float val = in[t] + bg[c];
  val = (val - m[c]) * rsqrtf(v[c] + 1e-5f) * g[c] + b[c];
  val = val > 0.f ? val : (expf(val) - 1.f);
  hOut[t] = val;
}

// ------------------------- pooling -------------------------
__global__ void k_pool(const float* __restrict__ h, const int* __restrict__ batch,
                       float* __restrict__ psum, unsigned* __restrict__ pmax,
                       float* __restrict__ pcnt, int N) {
  size_t t = (size_t)blockIdx.x * blockDim.x + threadIdx.x;
  if (t >= (size_t)N * CCH) return;
  int n = (int)(t >> 6), c = (int)(t & 63);
  int g = batch[n];
  float v = h[t];
  atomicAdd(psum + (size_t)g * CCH + c, v);
  atomicMax(pmax + (size_t)g * CCH + c, fenc(v));
  if (c == 0) atomicAdd(pcnt + g, 1.0f);
}
__global__ void k_xg(const float* __restrict__ psum, const unsigned* __restrict__ pmax,
                     const float* __restrict__ pcnt, const float* __restrict__ Wr,
                     const float* __restrict__ br, float* __restrict__ xg1,
                     float* __restrict__ xg2) {
  size_t t = (size_t)blockIdx.x * blockDim.x + threadIdx.x;
  if (t >= (size_t)GRP * CCH) return;
  int g = (int)(t >> 6), j = (int)(t & 63);
  float cnt = pcnt[g];
  float ic = 1.0f / fmaxf(cnt, 1.0f);
  float acc = br[j];
#pragma unroll 4
  for (int i = 0; i < CCH; ++i) acc += psum[(size_t)g * CCH + i] * ic * Wr[i * CCH + j];
#pragma unroll 4
  for (int i = 0; i < CCH; ++i) {
    float mx = cnt > 0.f ? fdec(pmax[(size_t)g * CCH + i]) : 0.f;
    acc += mx * Wr[(CCH + i) * CCH + j];
  }
  xg1[t] = acc; xg2[t] = acc;
}

// ------------------------- heads: one thread per graph -------------------------
__global__ void k_heads(const float* __restrict__ xg,
                        const float* Wh1, const float* bh1, const float* Wh2, const float* bh2,
                        const float* Wh3, const float* bh3,
                        const float* Wl1, const float* bl1, const float* Wl2, const float* bl2,
                        const float* We1, const float* be1, const float* We2, const float* be2,
                        float* out_rul, float* out_loc, float* out_en) {
  int g = (int)threadIdx.x;
  if (g >= GRP) return;
  const float* xr = xg + (size_t)g * CCH;
  float t1[32], t2[16];
  for (int j = 0; j < 32; ++j) {
    float a = bh1[j];
    for (int i = 0; i < CCH; ++i) a += xr[i] * Wh1[i * 32 + j];
    t1[j] = fmaxf(a, 0.f);
  }
  for (int j = 0; j < 16; ++j) {
    float a = bh2[j];
    for (int i = 0; i < 32; ++i) a += t1[i] * Wh2[i * 16 + j];
    t2[j] = fmaxf(a, 0.f);
  }
  float r = bh3[0];
  for (int i = 0; i < 16; ++i) r += t2[i] * Wh3[i];
  out_rul[g] = 100.0f / (1.0f + expf(-r));
  float l1[32];
  for (int j = 0; j < 32; ++j) {
    float a = bl1[j];
    for (int i = 0; i < CCH; ++i) a += xr[i] * Wl1[i * 32 + j];
    l1[j] = fmaxf(a, 0.f);
  }
  for (int j = 0; j < NCLS; ++j) {
    float a = bl2[j];
    for (int i = 0; i < 32; ++i) a += l1[i] * Wl2[i * NCLS + j];
    out_loc[(size_t)g * NCLS + j] = a;
  }
  float e1[16];
  for (int j = 0; j < 16; ++j) {
    float a = be1[j];
    for (int i = 0; i < CCH; ++i) a += xr[i] * We1[i * 16 + j];
    e1[j] = fmaxf(a, 0.f);
  }
  float en = be2[0];
  for (int i = 0; i < 16; ++i) en += e1[i] * We2[i];
  out_en[g] = en;
}

// ============================= launch =============================
extern "C" void kernel_launch(void* const* d_in, const int* in_sizes, int n_in,
                              void* d_out, int out_size, void* d_ws, size_t ws_size,
                              hipStream_t stream) {
  (void)n_in; (void)out_size; (void)ws_size;
  const float* x    = (const float*)d_in[0];
  const int*   ei   = (const int*)d_in[1];
  const int*   batch= (const int*)d_in[2];
  const float* W1   = (const float*)d_in[3];
  const float* as1  = (const float*)d_in[4];
  const float* ad1  = (const float*)d_in[5];
  const float* bi1  = (const float*)d_in[6];
  const float* W2   = (const float*)d_in[7];
  const float* as2  = (const float*)d_in[8];
  const float* ad2  = (const float*)d_in[9];
  const float* bi2  = (const float*)d_in[10];
  const float* bn1g = (const float*)d_in[11]; const float* bn1b = (const float*)d_in[12];
  const float* bn1m = (const float*)d_in[13]; const float* bn1v = (const float*)d_in[14];
  const float* bn2g = (const float*)d_in[15]; const float* bn2b = (const float*)d_in[16];
  const float* bn2m = (const float*)d_in[17]; const float* bn2v = (const float*)d_in[18];
  const float* bn3g = (const float*)d_in[19]; const float* bn3b = (const float*)d_in[20];
  const float* bn3m = (const float*)d_in[21]; const float* bn3v = (const float*)d_in[22];
  const float* Wg   = (const float*)d_in[23]; const float* bg   = (const float*)d_in[24];
  const float* Wr   = (const float*)d_in[25]; const float* br   = (const float*)d_in[26];
  const float* Wh1  = (const float*)d_in[27]; const float* bh1  = (const float*)d_in[28];
  const float* Wh2  = (const float*)d_in[29]; const float* bh2  = (const float*)d_in[30];
  const float* Wh3  = (const float*)d_in[31]; const float* bh3  = (const float*)d_in[32];
  const float* Wl1  = (const float*)d_in[33]; const float* bl1  = (const float*)d_in[34];
  const float* Wl2  = (const float*)d_in[35]; const float* bl2  = (const float*)d_in[36];
  const float* We1  = (const float*)d_in[37]; const float* be1  = (const float*)d_in[38];
  const float* We2  = (const float*)d_in[39]; const float* be2  = (const float*)d_in[40];

  const int N  = in_sizes[0] / FIN;
  const int E  = in_sizes[1] / 2;
  const int E2 = E + N;

  // ---- workspace carve-out (256B aligned) ----
  char* base = (char*)d_ws; size_t off = 0;
  auto take = [&](size_t bytes) -> void* {
    void* p = base + off; off = (off + bytes + 255) & ~(size_t)255; return p;
  };
  float*    hbuf  = (float*)take((size_t)N * HCD * 4);     // layer features
  float*    aggr  = (float*)take((size_t)N * HCD * 4);     // scatter target (reused 64-wide for GCN)
  float*    es    = (float*)take((size_t)N * HEADS * 4);
  float*    edv   = (float*)take((size_t)N * HEADS * 4);
  float*    eedge = (float*)take((size_t)E2 * HEADS * 4);
  unsigned* mEnc  = (unsigned*)take((size_t)N * HEADS * 4);
  float*    ssum  = (float*)take((size_t)N * HEADS * 4);
  _Float16* ahalf = (_Float16*)take((size_t)N * HCD * 2);
  _Float16* bthlf = (_Float16*)take((size_t)HCD * HCD * 2);
  float*    xw    = (float*)take((size_t)N * CCH * 4);
  float*    deg   = (float*)take((size_t)N * 4);
  float*    dinv  = (float*)take((size_t)N * 4);
  float*    psum  = (float*)take((size_t)GRP * CCH * 4);
  unsigned* pmax  = (unsigned*)take((size_t)GRP * CCH * 4);
  float*    pcnt  = (float*)take((size_t)GRP * 4);
  float*    xgws  = (float*)take((size_t)GRP * CCH * 4);

  float* out_rul = (float*)d_out;
  float* out_loc = out_rul + GRP;
  float* out_en  = out_rul + GRP + GRP * NCLS;
  float* out_h   = out_rul + GRP * (1 + NCLS + 1);
  float* out_xg  = out_h + (size_t)N * CCH;

  const int B = 256;
  auto nb = [&](size_t n) { return dim3((unsigned)((n + B - 1) / B)); };

  auto gat_edges = [&](const float* as_, const float* ad_) {
    k_esed<<<nb((size_t)N * HEADS), B, 0, stream>>>(hbuf, as_, ad_, es, edv, N);
    k_fill_u32<<<nb((size_t)N * HEADS), B, 0, stream>>>(mEnc, FENC_NEG_INF, (size_t)N * HEADS);
    k_fill_f32<<<nb((size_t)N * HEADS), B, 0, stream>>>(ssum, 0.f, (size_t)N * HEADS);
    k_fill_f32<<<nb((size_t)N * HCD), B, 0, stream>>>(aggr, 0.f, (size_t)N * HCD);
    k_edge_max<<<nb((size_t)E2 * HEADS), B, 0, stream>>>(ei, E, E2, es, edv, eedge, mEnc);
    k_edge_exp<<<nb((size_t)E2 * HEADS), B, 0, stream>>>(ei, E, E2, eedge, mEnc, ssum);
    k_edge_aggr<<<nb((size_t)E2 * HCD), B, 0, stream>>>(ei, E, E2, eedge, ssum, hbuf, aggr);
  };

  // ---- GAT layer 1 ----
  k_lin1<<<nb((size_t)N * HCD), B, 0, stream>>>(x, W1, hbuf, N);
  gat_edges(as1, ad1);
  k_post<<<nb((size_t)N * HCD), B, 0, stream>>>(aggr, bi1, bn1g, bn1b, bn1m, bn1v, hbuf, ahalf, N);

  // ---- GAT layer 2: h2 = act1 @ W2 via WMMA f16->f32 (16x64 strip per wave) ----
  k_wT<<<nb((size_t)HCD * HCD), B, 0, stream>>>(W2, bthlf, HCD, HCD);
  {
    size_t waves = (size_t)(N / 16) * (HCD / 64);
    k_gemm_wmma<<<nb(waves * 32), B, 0, stream>>>(ahalf, bthlf, hbuf, N, HCD, HCD);
  }
  gat_edges(as2, ad2);
  k_post<<<nb((size_t)N * HCD), B, 0, stream>>>(aggr, bi2, bn2g, bn2b, bn2m, bn2v, hbuf, ahalf, N);

  // ---- GCN: xw = act2 @ Wg via WMMA; normalized scatter ----
  k_wT<<<nb((size_t)HCD * CCH), B, 0, stream>>>(Wg, bthlf, HCD, CCH);
  {
    size_t waves = (size_t)(N / 16) * (CCH / 64);
    k_gemm_wmma<<<nb(waves * 32), B, 0, stream>>>(ahalf, bthlf, xw, N, HCD, CCH);
  }
  k_fill_f32<<<nb((size_t)N), B, 0, stream>>>(deg, 0.f, (size_t)N);
  k_deg<<<nb((size_t)E2), B, 0, stream>>>(ei, E, E2, deg);
  k_dinv<<<nb((size_t)N), B, 0, stream>>>(deg, dinv, N);
  k_fill_f32<<<nb((size_t)N * CCH), B, 0, stream>>>(aggr, 0.f, (size_t)N * CCH);
  k_gcn_edge<<<nb((size_t)E2 * CCH), B, 0, stream>>>(ei, E, E2, xw, dinv, aggr);
  k_post3<<<nb((size_t)N * CCH), B, 0, stream>>>(aggr, bg, bn3g, bn3b, bn3m, bn3v, out_h, N);

  // ---- pooling + reduce_dim ----
  k_fill_f32<<<nb((size_t)GRP * CCH), B, 0, stream>>>(psum, 0.f, (size_t)GRP * CCH);
  k_fill_u32<<<nb((size_t)GRP * CCH), B, 0, stream>>>(pmax, FENC_NEG_INF, (size_t)GRP * CCH);
  k_fill_f32<<<nb((size_t)GRP), B, 0, stream>>>(pcnt, 0.f, (size_t)GRP);
  k_pool<<<nb((size_t)N * CCH), B, 0, stream>>>(out_h, batch, psum, pmax, pcnt, N);
  k_xg<<<nb((size_t)GRP * CCH), B, 0, stream>>>(psum, pmax, pcnt, Wr, br, xgws, out_xg);

  // ---- heads ----
  k_heads<<<1, GRP, 0, stream>>>(xgws, Wh1, bh1, Wh2, bh2, Wh3, bh3,
                                 Wl1, bl1, Wl2, bl2, We1, be1, We2, be2,
                                 out_rul, out_loc, out_en);
}
